// ReinforcedDecoder_14998025797911
// MI455X (gfx1250) — matrix operand order repaired and checked by reading, not compiled, not run on hardware
//
#include <hip/hip_runtime.h>
#include <cstdint>
#include <cstddef>

// ---------------- problem constants ----------------
#define T_STEPS 64
#define BSZ     64
#define DIM     1024
#define EMBD    512
#define SRC_N   512
#define G4      (4 * DIM)
#define NEGINF  (-1e30f)

// ---------------- vector types ----------------
typedef __attribute__((ext_vector_type(16))) __bf16 v16bf;
typedef __attribute__((ext_vector_type(8)))  float  v8f;
typedef __attribute__((ext_vector_type(4)))  __bf16 bf4;
typedef __attribute__((ext_vector_type(4)))  float  f4;
typedef __attribute__((ext_vector_type(4)))  unsigned int u4;
typedef __attribute__((ext_vector_type(4)))  int    v4i_t;

union Frag16 { v16bf v; u4 q[2]; };

// ---------------- CDNA5 async global->LDS copy (guarded) -------------------
#if defined(__HIP_DEVICE_COMPILE__) && __has_builtin(__builtin_amdgcn_global_load_async_to_lds_b128)
#define HAVE_ASYNC_LDS 1
#else
#define HAVE_ASYNC_LDS 0
#endif

#if HAVE_ASYNC_LDS
typedef __attribute__((address_space(1))) v4i_t* global_v4i_ptr;
typedef __attribute__((address_space(3))) v4i_t* lds_v4i_ptr;
#endif

__device__ __forceinline__ void copy16_g2l(const void* g, void* l) {
#if HAVE_ASYNC_LDS
  __builtin_amdgcn_global_load_async_to_lds_b128(
      (global_v4i_ptr)g, (lds_v4i_ptr)l, 0, 0);
#else
  *reinterpret_cast<u4*>(l) = *reinterpret_cast<const u4*>(g);
#endif
}

__device__ __forceinline__ void wait_async_all() {
#if HAVE_ASYNC_LDS
#if __has_builtin(__builtin_amdgcn_s_wait_asynccnt)
  __builtin_amdgcn_s_wait_asynccnt(0);
#else
  asm volatile("s_wait_asynccnt 0x0" ::: "memory");
#endif
#endif
}

// ---------------- WMMA fragment loaders (per CDNA5 ISA 7.12.2 layouts) ----
// A: 16x32 bf16, row-major [M x K] source (works for global or LDS pointers).
// lane&15 = M row; per lane two contiguous 8-half runs: K=kbase+hi*8 and +16.
__device__ __forceinline__ v16bf load_a_frag(const __bf16* __restrict__ A,
                                             int ldk, int mbase, int kbase, int lane) {
  const int m  = mbase + (lane & 15);
  const int hi = lane >> 4;
  Frag16 f;
  f.q[0] = *reinterpret_cast<const u4*>(A + (size_t)m * ldk + kbase + hi * 8);
  f.q[1] = *reinterpret_cast<const u4*>(A + (size_t)m * ldk + kbase + 16 + hi * 8);
  return f.v;
}

// B: 32x16 bf16 from torch weight W[N x K] row-major (B[k][n] = W[n][k]).
// lane&15 = N column; per lane two contiguous 8-half runs: K=kbase+hi*8 and +16.
__device__ __forceinline__ v16bf load_b_frag(const __bf16* __restrict__ W,
                                             int ldk, int nbase, int kbase, int lane) {
  const int n = nbase + (lane & 15);
  const int k = kbase + ((lane >> 4) << 3);
  Frag16 f;
  const u4* p = reinterpret_cast<const u4*>(W + (size_t)n * ldk + k);
  f.q[0] = p[0];
  f.q[1] = p[2];
  return f.v;
}

// ---------------- conversion / init kernels ----------------
__global__ __launch_bounds__(256) void cvt_bf16_kernel(const float* __restrict__ src,
                                                       __bf16* __restrict__ dst, size_t n) {
  size_t i = (size_t)blockIdx.x * blockDim.x + threadIdx.x;
  const size_t stride = (size_t)gridDim.x * blockDim.x;
  for (; i < n; i += stride) dst[i] = (__bf16)src[i];
}

__global__ __launch_bounds__(256) void zero_f32_kernel(float* __restrict__ dst, size_t n) {
  size_t i = (size_t)blockIdx.x * blockDim.x + threadIdx.x;
  const size_t stride = (size_t)gridDim.x * blockDim.x;
  for (; i < n; i += stride) dst[i] = 0.f;
}

__global__ __launch_bounds__(256) void init_ms_kernel(float* __restrict__ m,
                                                      float* __restrict__ s) {
  const int i = blockIdx.x * blockDim.x + threadIdx.x;
  if (i < BSZ * SRC_N) { m[i] = NEGINF; s[i] = 0.f; }
}

__global__ __launch_bounds__(256) void init_hc_kernel(const float* __restrict__ h0,
                                                      const float* __restrict__ c0,
                                                      float* __restrict__ h,
                                                      float* __restrict__ c,
                                                      __bf16* __restrict__ hbf) {
  const int i = blockIdx.x * blockDim.x + threadIdx.x;
  if (i < BSZ * DIM) {
    const float hv = h0[i];
    h[i] = hv; c[i] = c0[i]; hbf[i] = (__bf16)hv;
  }
}

// ---------------- embedding gather + bf16 convert ----------------
__global__ __launch_bounds__(256) void embed_kernel(const int* __restrict__ tok_t,
                                                    const float* __restrict__ W_emb,
                                                    __bf16* __restrict__ emb_bf) {
  const int i = blockIdx.x * blockDim.x + threadIdx.x;  // BSZ*EMBD
  const int b = i >> 9, k = i & (EMBD - 1);
  const int tok = tok_t[b];
  emb_bf[i] = (__bf16)W_emb[(size_t)tok * EMBD + k];
}

// ---------------- fused LSTM gate GEMM: G = emb@Wih^T + h@Whh^T + bias -----
// Activations (A, 192KB) async-staged to LDS once per block; one wave owns
// 1 N-tile x 4 M-tiles so each B fragment is loaded once and reused 4x.
// 32 blocks x 8 waves cover N=4096.
__global__ __launch_bounds__(256) void gates_gemm_kernel(
    const __bf16* __restrict__ emb, const __bf16* __restrict__ hbf,
    const __bf16* __restrict__ Wih, const __bf16* __restrict__ Whh,
    const float* __restrict__ bih, const float* __restrict__ bhh,
    float* __restrict__ G) {
  extern __shared__ __align__(16) char smem[];   // 192KB: emb[64][512] | h[64][1024]
  const int tid = threadIdx.x;
  for (int i = tid; i < 12288; i += 256) {       // 12288 x 16B chunks
    const char* src = (i < 4096) ? ((const char*)emb + (size_t)i * 16)
                                 : ((const char*)hbf + (size_t)(i - 4096) * 16);
    copy16_g2l(src, smem + (size_t)i * 16);
  }
  wait_async_all();
  __syncthreads();
  const __bf16* sEmb = (const __bf16*)smem;
  const __bf16* sH   = (const __bf16*)smem + BSZ * EMBD;

  const int lane  = tid & 31;
  const int nbase = ((blockIdx.x << 3) | (tid >> 5)) << 4;  // 256 N-tiles
  v8f acc[4] = {};
  for (int k = 0; k < EMBD; k += 32) {
    const v16bf bfrg = load_b_frag(Wih, EMBD, nbase, k, lane);
#pragma unroll
    for (int mt = 0; mt < 4; ++mt) {
      const v16bf afrg = load_a_frag(sEmb, EMBD, mt * 16, k, lane);
      acc[mt] = __builtin_amdgcn_wmma_f32_16x16x32_bf16(false, afrg, false, bfrg,
                                                        (short)0, acc[mt], false, false);
    }
  }
  for (int k = 0; k < DIM; k += 32) {
    const v16bf bfrg = load_b_frag(Whh, DIM, nbase, k, lane);
#pragma unroll
    for (int mt = 0; mt < 4; ++mt) {
      const v16bf afrg = load_a_frag(sH, DIM, mt * 16, k, lane);
      acc[mt] = __builtin_amdgcn_wmma_f32_16x16x32_bf16(false, afrg, false, bfrg,
                                                        (short)0, acc[mt], false, false);
    }
  }
  const int n = nbase + (lane & 15);
  const float bias = bih[n] + bhh[n];
#pragma unroll
  for (int mt = 0; mt < 4; ++mt) {
    const int mrow = mt * 16 + ((lane >> 4) << 3);
#pragma unroll
    for (int r = 0; r < 8; ++r)
      G[(size_t)(mrow + r) * G4 + n] = acc[mt][r] + bias;
  }
}

// ---------------- merged projections: q = h@Wenc^T+b, qd = h@Wdec^T+b ------
// h (128KB) async-staged to LDS; 16 blocks x 8 waves cover both outputs.
__global__ __launch_bounds__(256) void proj2_gemm_kernel(
    const __bf16* __restrict__ hbf,
    const __bf16* __restrict__ Wenc, const __bf16* __restrict__ Wdec,
    const float* __restrict__ benc, const float* __restrict__ bdec,
    float* __restrict__ q, float* __restrict__ qd) {
  extern __shared__ __align__(16) char smem[];   // 128KB: h[64][1024]
  const int tid = threadIdx.x;
  for (int i = tid; i < 8192; i += 256)
    copy16_g2l((const char*)hbf + (size_t)i * 16, smem + (size_t)i * 16);
  wait_async_all();
  __syncthreads();
  const __bf16* sH = (const __bf16*)smem;

  const int lane = tid & 31;
  const int nb   = (blockIdx.x << 3) | (tid >> 5);  // 0..127 (2 x 64 N-tiles)
  const bool dec = nb >= 64;
  const __bf16* W    = dec ? Wdec : Wenc;
  const float*  bias = dec ? bdec : benc;
  float*        C    = dec ? qd : q;
  const int nbase = (nb & 63) << 4;
  v8f acc[4] = {};
  for (int k = 0; k < DIM; k += 32) {
    const v16bf bfrg = load_b_frag(W, DIM, nbase, k, lane);
#pragma unroll
    for (int mt = 0; mt < 4; ++mt) {
      const v16bf afrg = load_a_frag(sH, DIM, mt * 16, k, lane);
      acc[mt] = __builtin_amdgcn_wmma_f32_16x16x32_bf16(false, afrg, false, bfrg,
                                                        (short)0, acc[mt], false, false);
    }
  }
  const int n = nbase + (lane & 15);
  const float bv = bias[n];
#pragma unroll
  for (int mt = 0; mt < 4; ++mt) {
    const int mrow = mt * 16 + ((lane >> 4) << 3);
#pragma unroll
    for (int r = 0; r < 8; ++r)
      C[(size_t)(mrow + r) * DIM + n] = acc[mt][r] + bv;
  }
}

// ---------------- LSTM pointwise ----------------
__global__ __launch_bounds__(256) void lstm_cell_kernel(const float* __restrict__ G,
                                                        float* __restrict__ h,
                                                        float* __restrict__ c,
                                                        __bf16* __restrict__ hbf) {
  const int i = blockIdx.x * blockDim.x + threadIdx.x;  // BSZ*DIM
  const int b = i >> 10, d = i & (DIM - 1);
  const float gi = G[(size_t)b * G4 + d];
  const float gf = G[(size_t)b * G4 + DIM + d];
  const float gg = G[(size_t)b * G4 + 2 * DIM + d];
  const float go = G[(size_t)b * G4 + 3 * DIM + d];
  const float si = 1.f / (1.f + __expf(-gi));
  const float sf = 1.f / (1.f + __expf(-gf));
  const float so = 1.f / (1.f + __expf(-go));
  const float c2 = sf * c[i] + si * tanhf(gg);
  const float h2 = so * tanhf(c2);
  c[i] = c2; h[i] = h2; hbf[i] = (__bf16)h2;
}

// ---------------- fused encoder intra-attention (single pass over h_e) -----
// One block per batch row b. Each warp flash-accumulates (max, sum, context)
// over its 64 source positions, reusing each loaded h_e row for both the
// score dot-product and the context accumulation; warp partials merged in LDS.
// Also performs the across-time online-softmax state update of (m, s).
__global__ __launch_bounds__(256) void enc_attn_kernel(
    const float* __restrict__ q, const __bf16* __restrict__ he,
    float* __restrict__ mstate, float* __restrict__ sstate,
    float* __restrict__ out, int t) {
  const int b = blockIdx.x, tid = threadIdx.x;
  const int lane = tid & 31, w = tid >> 5;
  __shared__ __align__(16) float sq[DIM];
  __shared__ __align__(16) float sctx[8][DIM];
  __shared__ float smw[8], ssw[8];
  for (int d = tid; d < DIM; d += 256) sq[d] = q[(size_t)b * DIM + d];
  __syncthreads();

  float mw = NEGINF, sw = 0.f;
  const f4 zero4 = {0.f, 0.f, 0.f, 0.f};
  f4 ctx[8];
#pragma unroll
  for (int j = 0; j < 8; ++j) ctx[j] = zero4;

  for (int n = w; n < SRC_N; n += 8) {
    const __bf16* hp = he + ((size_t)n * BSZ + b) * DIM;
    bf4 hv[8];
    float dot = 0.f;
#pragma unroll
    for (int j = 0; j < 8; ++j) {
      const int d0 = j * 128 + lane * 4;
      hv[j] = *reinterpret_cast<const bf4*>(hp + d0);
      const f4 qv = *reinterpret_cast<const f4*>(sq + d0);
      dot += qv[0] * (float)hv[j][0] + qv[1] * (float)hv[j][1] +
             qv[2] * (float)hv[j][2] + qv[3] * (float)hv[j][3];
    }
#pragma unroll
    for (int off = 16; off; off >>= 1) dot += __shfl_xor(dot, off, 32);
    // across-time online softmax state (per (b, n) element)
    const int idx = b * SRC_N + n;
    const float mo = mstate[idx], so = sstate[idx];
    const float m2 = fmaxf(mo, dot);
    const float s2 = so * __expf(mo - m2) + __expf(dot - m2);
    if (lane == 0) { mstate[idx] = m2; sstate[idx] = s2; }
    const float sc = (t == 0) ? dot : (__expf(dot - m2) / s2);
    // warp-local flash accumulation of softmax(sc) . h_e
    const float mnew = fmaxf(mw, sc);
    const float f    = __expf(mw - mnew);
    const float wgt  = __expf(sc - mnew);
    sw = sw * f + wgt;
    mw = mnew;
#pragma unroll
    for (int j = 0; j < 8; ++j) {
      ctx[j][0] = ctx[j][0] * f + wgt * (float)hv[j][0];
      ctx[j][1] = ctx[j][1] * f + wgt * (float)hv[j][1];
      ctx[j][2] = ctx[j][2] * f + wgt * (float)hv[j][2];
      ctx[j][3] = ctx[j][3] * f + wgt * (float)hv[j][3];
    }
  }
  if (lane == 0) { smw[w] = mw; ssw[w] = sw; }
#pragma unroll
  for (int j = 0; j < 8; ++j)
    *reinterpret_cast<f4*>(&sctx[w][j * 128 + lane * 4]) = ctx[j];
  __syncthreads();
  // merge the 8 warp partials
  float M = smw[0];
#pragma unroll
  for (int i = 1; i < 8; ++i) M = fmaxf(M, smw[i]);
  float S = 0.f;
#pragma unroll
  for (int i = 0; i < 8; ++i) S += ssw[i] * __expf(smw[i] - M);
  float* op = out + ((size_t)t * BSZ + b) * (3 * DIM) + DIM;
  for (int d = tid; d < DIM; d += 256) {
    float acc = 0.f;
#pragma unroll
    for (int i = 0; i < 8; ++i) acc += sctx[i][d] * __expf(smw[i] - M);
    op[d] = acc / S;
  }
}

// ---------------- decoder intra-attention + buf append + output h/cd -------
__global__ __launch_bounds__(256) void dec_attn_kernel(const float* __restrict__ qd,
                                                       const float* __restrict__ h,
                                                       float* __restrict__ buf,
                                                       float* __restrict__ out, int t) {
  const int b = blockIdx.x, tid = threadIdx.x;
  const int lane = tid & 31, w = tid >> 5;
  const int len = (t > 0) ? t : 1;
  __shared__ float ed[T_STEPS];
  __shared__ float alpha[T_STEPS];
  const float* qb = qd + (size_t)b * DIM;
  for (int tt = w; tt < len; tt += 8) {
    const float* bp = buf + ((size_t)tt * BSZ + b) * DIM;
    float s = 0.f;
    for (int d = lane; d < DIM; d += 32) s += qb[d] * bp[d];
#pragma unroll
    for (int off = 16; off; off >>= 1) s += __shfl_xor(s, off, 32);
    if (!lane) ed[tt] = s;
  }
  __syncthreads();
  if (w == 0) {
    float v0 = (lane < len) ? ed[lane] : NEGINF;
    float v1 = (lane + 32 < len) ? ed[lane + 32] : NEGINF;
    float mx = fmaxf(v0, v1);
#pragma unroll
    for (int off = 16; off; off >>= 1) mx = fmaxf(mx, __shfl_xor(mx, off, 32));
    float e0 = (lane < len) ? __expf(v0 - mx) : 0.f;
    float e1 = (lane + 32 < len) ? __expf(v1 - mx) : 0.f;
    float sm = e0 + e1;
#pragma unroll
    for (int off = 16; off; off >>= 1) sm += __shfl_xor(sm, off, 32);
    if (lane < len) alpha[lane] = e0 / sm;
    if (lane + 32 < len) alpha[lane + 32] = e1 / sm;
  }
  __syncthreads();
  float* ob = out + ((size_t)t * BSZ + b) * (3 * DIM);
  for (int d = tid; d < DIM; d += 256) {
    float acc = 0.f;
    for (int tt = 0; tt < len; ++tt) acc += alpha[tt] * buf[((size_t)tt * BSZ + b) * DIM + d];
    ob[2 * DIM + d] = (t > 0) ? acc : 0.f;
    const float hv = h[(size_t)b * DIM + d];
    ob[d] = hv;
    buf[((size_t)t * BSZ + b) * DIM + d] = hv;  // append after use
  }
}

// ---------------- host launcher ----------------
extern "C" void kernel_launch(void* const* d_in, const int* in_sizes, int n_in,
                              void* d_out, int out_size, void* d_ws, size_t ws_size,
                              hipStream_t stream) {
  const int*   inputs = (const int*)d_in[0];
  const float* h_e    = (const float*)d_in[1];
  const float* h0     = (const float*)d_in[2];
  const float* c0     = (const float*)d_in[3];
  const float* W_emb  = (const float*)d_in[4];
  const float* W_ih   = (const float*)d_in[5];
  const float* W_hh   = (const float*)d_in[6];
  const float* b_ih   = (const float*)d_in[7];
  const float* b_hh   = (const float*)d_in[8];
  const float* W_enc  = (const float*)d_in[9];
  const float* b_enc  = (const float*)d_in[10];
  const float* W_dec  = (const float*)d_in[11];
  const float* b_dec  = (const float*)d_in[12];
  float* out = (float*)d_out;

  // workspace carve-out (256B aligned slabs)
  char* wp = (char*)d_ws;
  auto take = [&](size_t bytes) -> void* {
    void* p = wp;
    wp += (bytes + 255) & ~(size_t)255;
    return p;
  };
  __bf16* he_bf   = (__bf16*)take((size_t)SRC_N * BSZ * DIM * 2);
  __bf16* Wih_bf  = (__bf16*)take((size_t)G4 * EMBD * 2);
  __bf16* Whh_bf  = (__bf16*)take((size_t)G4 * DIM * 2);
  __bf16* Wenc_bf = (__bf16*)take((size_t)DIM * DIM * 2);
  __bf16* Wdec_bf = (__bf16*)take((size_t)DIM * DIM * 2);
  __bf16* emb_bf  = (__bf16*)take((size_t)BSZ * EMBD * 2);
  __bf16* h_bf    = (__bf16*)take((size_t)BSZ * DIM * 2);
  float*  h       = (float*)take((size_t)BSZ * DIM * 4);
  float*  c       = (float*)take((size_t)BSZ * DIM * 4);
  float*  G       = (float*)take((size_t)BSZ * G4 * 4);
  float*  q       = (float*)take((size_t)BSZ * DIM * 4);
  float*  qdv     = (float*)take((size_t)BSZ * DIM * 4);
  float*  m       = (float*)take((size_t)BSZ * SRC_N * 4);
  float*  s       = (float*)take((size_t)BSZ * SRC_N * 4);
  float*  buf     = (float*)take((size_t)T_STEPS * BSZ * DIM * 4);
  (void)ws_size; (void)n_in; (void)in_sizes; (void)out_size;

  // allow the large dynamic-LDS allocations (no-op if unnecessary)
  (void)hipFuncSetAttribute((const void*)gates_gemm_kernel,
                            hipFuncAttributeMaxDynamicSharedMemorySize, 196608);
  (void)hipFuncSetAttribute((const void*)proj2_gemm_kernel,
                            hipFuncAttributeMaxDynamicSharedMemorySize, 131072);

  // one-time (per launch) conversions — h_e drops to 67MB bf16, L2-resident
  cvt_bf16_kernel<<<4096, 256, 0, stream>>>(h_e, he_bf, (size_t)SRC_N * BSZ * DIM);
  cvt_bf16_kernel<<<1024, 256, 0, stream>>>(W_ih, Wih_bf, (size_t)G4 * EMBD);
  cvt_bf16_kernel<<<2048, 256, 0, stream>>>(W_hh, Whh_bf, (size_t)G4 * DIM);
  cvt_bf16_kernel<<<512, 256, 0, stream>>>(W_enc, Wenc_bf, (size_t)DIM * DIM);
  cvt_bf16_kernel<<<512, 256, 0, stream>>>(W_dec, Wdec_bf, (size_t)DIM * DIM);
  zero_f32_kernel<<<4096, 256, 0, stream>>>(buf, (size_t)T_STEPS * BSZ * DIM);
  init_ms_kernel<<<(BSZ * SRC_N) / 256, 256, 0, stream>>>(m, s);
  init_hc_kernel<<<(BSZ * DIM) / 256, 256, 0, stream>>>(h0, c0, h, c, h_bf);

  for (int t = 0; t < T_STEPS; ++t) {
    embed_kernel<<<(BSZ * EMBD) / 256, 256, 0, stream>>>(inputs + t * BSZ, W_emb, emb_bf);
    gates_gemm_kernel<<<32, 256, 196608, stream>>>(emb_bf, h_bf, Wih_bf, Whh_bf,
                                                   b_ih, b_hh, G);
    lstm_cell_kernel<<<(BSZ * DIM) / 256, 256, 0, stream>>>(G, h, c, h_bf);
    proj2_gemm_kernel<<<16, 256, 131072, stream>>>(h_bf, Wenc_bf, Wdec_bf,
                                                   b_enc, b_dec, q, qdv);
    enc_attn_kernel<<<BSZ, 256, 0, stream>>>(q, he_bf, m, s, out, t);
    dec_attn_kernel<<<BSZ, 256, 0, stream>>>(qdv, h, buf, out, t);
  }
}
